// Encoder_10110353014809
// MI455X (gfx1250) — compile-verified, matrix operand order checked by reading, not run on hardware
//
#include <hip/hip_runtime.h>
#include <hip/hip_bf16.h>
#include <math.h>

// ---------------------------------------------------------------------------
// Types
// ---------------------------------------------------------------------------
typedef __attribute__((ext_vector_type(16))) __bf16   v16bf;
typedef __attribute__((ext_vector_type(8)))  __bf16   v8bf;
typedef __attribute__((ext_vector_type(8)))  float    v8f;
typedef __attribute__((ext_vector_type(4)))  unsigned v4u;

#define MODE_ZERO    0   // zero padding, arbitrary stride
#define MODE_REFLECT 1   // reflect padding (stride 1)
#define MODE_TCONV   2   // ConvTranspose2d(k=3,s=2,p=1,op=1) as lhs-dilated conv

#define NUM_INST 32
#define HW_OUT   262144          // 512*512 (final image HW)

__device__ __forceinline__ int reflect_idx(int i, int n) {
    if (i < 0)  i = -i;
    if (i >= n) i = 2 * n - 2 - i;
    return i;
}

__device__ __forceinline__ int klocal_of(int e, int group) {
    // 16-bit A/B fragment K interleave (ISA 7.12.2)
    return (e < 8) ? (group * 8 + e) : (8 + group * 8 + e);
}

// ---------------------------------------------------------------------------
// Weight repack -> bf16 in exact B-fragment order (one 32B load per chunk).
// Chunk order q = (kh*KW + kw)*(Cin/32) + cb/32 matches the conv loop nest.
// ---------------------------------------------------------------------------
__global__ __launch_bounds__(256)
void pack_w_cin32(const float* __restrict__ w, __bf16* __restrict__ out,
                  int Cin, int Cout, int KH, int KW, int trans, int tilesN,
                  int total)
{
    const int idx = blockIdx.x * blockDim.x + threadIdx.x;
    if (idx >= total) return;
    const int e    = idx & 15;
    const int lane = (idx >> 4) & 31;
    const int rest = idx >> 9;
    const int tn   = rest % tilesN;
    const int q    = rest / tilesN;
    const int cch  = Cin >> 5;
    const int cb   = (q % cch) << 5;
    const int tap  = q / cch;
    const int kh   = tap / KW;
    const int kw   = tap - kh * KW;
    const int ci   = cb + klocal_of(e, lane >> 4);
    int co = tn * 16 + (lane & 15);
    if (co >= Cout) co = Cout - 1;          // clamped; those columns never stored
    float v;
    if (trans)  // ref: flip(kh,kw) + swap in/out dims; w is [Cin,Cout,KH,KW]
        v = w[(((size_t)ci * Cout + co) * KH + (KH - 1 - kh)) * KW + (KW - 1 - kw)];
    else        // w is [Cout,Cin,KH,KW]
        v = w[(((size_t)co * Cin + ci) * KH + kh) * KW + kw];
    out[idx] = (__bf16)v;
}

// Head weights: Cin=3, 7x7, K = 147 padded to 160 (5 chunks), tilesN = 2.
__global__ __launch_bounds__(256)
void pack_w_head(const float* __restrict__ w, __bf16* __restrict__ out)
{
    const int idx = blockIdx.x * blockDim.x + threadIdx.x;
    if (idx >= 5 * 2 * 32 * 16) return;
    const int e    = idx & 15;
    const int lane = (idx >> 4) & 31;
    const int tn   = (idx >> 9) & 1;
    const int q    = idx >> 10;
    const int k    = q * 32 + klocal_of(e, lane >> 4);
    float v = 0.0f;
    if (k < 147) {
        const int tap = k / 3;              // K order: (kh,kw) outer, ci inner
        const int ci  = k - tap * 3;
        const int kh  = tap / 7;
        const int kw  = tap - kh * 7;
        const int co  = tn * 16 + (lane & 15);
        v = w[(((size_t)co * 3 + ci) * 7 + kh) * 7 + kw];
    }
    out[idx] = (__bf16)v;
}

// ---------------------------------------------------------------------------
// Implicit-GEMM conv, activations in bf16 NHWC, weights pre-packed bf16.
// Per K-chunk per lane: two 16B A loads + one 32B B load + v_wmma. Padding is
// a branch-free AND-mask on the loaded A vectors.
// ---------------------------------------------------------------------------
template<int MODE>
__global__ __launch_bounds__(256)
void wmma_conv_cin32(const __bf16* __restrict__ x, const __bf16* __restrict__ wp,
                     const float* __restrict__ bias, float* __restrict__ y,
                     int N, int Cin, int H, int W, int Cout, int Ho, int Wo,
                     int KH, int KW, int stride, int pad, int act)
{
    const int lane  = threadIdx.x & 31;
    const int wave  = threadIdx.x >> 5;
    const int group = lane >> 4;
    const int mrow  = lane & 15;

    const int tilesN  = (Cout + 15) >> 4;
    const int tilesWo = Wo >> 4;                  // Wo is a multiple of 16
    const int tilesM  = N * Ho * tilesWo;

    const int tile = blockIdx.x * 8 + wave;
    if (tile >= tilesM * tilesN) return;          // wave-uniform

    const int tn  = tile % tilesN;
    int tm        = tile / tilesN;
    const int owb = (tm % tilesWo) << 4;
    tm           /= tilesWo;
    const int oh  = tm % Ho;
    const int n   = tm / Ho;

    const int ow = owb + mrow;                    // this lane's A-row pixel
    const int co = tn * 16 + mrow;                // this lane's B-column

    const __bf16* xn = x + (size_t)n * H * W * Cin;   // NHWC

    // packed-B walking pointer: advances by bstep elements per K-chunk
    const __bf16* bptr = wp + ((size_t)tn * 32 + lane) * 16;
    const size_t bstep = (size_t)tilesN * 512;        // tilesN*32*16 elements

    v8f acc = {};

    for (int kh = 0; kh < KH; ++kh) {
        for (int kw = 0; kw < KW; ++kw) {
            // ---- per-tap input coordinate for this lane's pixel ----
            int ih, iw; bool ok;
            if (MODE == MODE_TCONV) {
                const int vr = oh + kh - 1;       // pad_low = 1, lhs_dilation 2
                const int vc = ow + kw - 1;
                ih = vr >> 1; iw = vc >> 1;
                ok = (vr >= 0) && ((vr & 1) == 0) && (ih < H) &&
                     (vc >= 0) && ((vc & 1) == 0) && (iw < W);
            } else if (MODE == MODE_REFLECT) {
                ih = reflect_idx(oh * stride + kh - pad, H);
                iw = reflect_idx(ow * stride + kw - pad, W);
                ok = true;
            } else {
                ih = oh * stride + kh - pad;
                iw = ow * stride + kw - pad;
                ok = (ih >= 0) && (ih < H) && (iw >= 0) && (iw < W);
            }
            const int ihc = ih < 0 ? 0 : (ih >= H ? H - 1 : ih);
            const int iwc = iw < 0 ? 0 : (iw >= W ? W - 1 : iw);
            const unsigned msk = ok ? 0xFFFFFFFFu : 0u;

            // NHWC: 16 K-values = two contiguous 8xbf16 runs per lane
            const __bf16* ap = xn + (size_t)(ihc * (size_t)W + iwc) * Cin
                                  + (group << 3);

            for (int cb = 0; cb < Cin; cb += 32) {
                const v16bf b = *(const v16bf*)bptr;
                bptr += bstep;

                v4u lu = *(const v4u*)(ap);        // ci = cb+g*8 .. +7
                v4u hu = *(const v4u*)(ap + 16);   // ci = cb+16+g*8 .. +7
                ap += 32;
                if (MODE != MODE_REFLECT) { lu &= msk; hu &= msk; }
                const v8bf lo = __builtin_bit_cast(v8bf, lu);
                const v8bf hi = __builtin_bit_cast(v8bf, hu);
                const v16bf a = __builtin_shufflevector(lo, hi,
                    0, 1, 2, 3, 4, 5, 6, 7, 8, 9, 10, 11, 12, 13, 14, 15);

                acc = __builtin_amdgcn_wmma_f32_16x16x32_bf16(
                          false, a, false, b, (short)0, acc, false, false);
            }
        }
    }

    // ---- epilogue: bias (+tanh), two 128-bit stores of 8 consecutive ow ----
    if (co < Cout) {
        const float bval = bias[co];
        float4 lo, hi;
        lo.x = acc[0] + bval; lo.y = acc[1] + bval;
        lo.z = acc[2] + bval; lo.w = acc[3] + bval;
        hi.x = acc[4] + bval; hi.y = acc[5] + bval;
        hi.z = acc[6] + bval; hi.w = acc[7] + bval;
        if (act == 1) {
            lo.x = tanhf(lo.x); lo.y = tanhf(lo.y);
            lo.z = tanhf(lo.z); lo.w = tanhf(lo.w);
            hi.x = tanhf(hi.x); hi.y = tanhf(hi.y);
            hi.z = tanhf(hi.z); hi.w = tanhf(hi.w);
        }
        float* yb = y + (((size_t)n * Cout + co) * Ho + oh) * Wo + owb + group * 8;
        *(float4*)(yb)     = lo;
        *(float4*)(yb + 4) = hi;
    }
}

// ---------------------------------------------------------------------------
// Head conv: 3->32, 7x7, reflect pad 3, 512x512, batch 2; reads f32 NCHW input
// (Cin=3 -> scalar gather is fine), packed bf16 weights, writes f32 NCHW.
// ---------------------------------------------------------------------------
__global__ __launch_bounds__(256)
void wmma_conv_head(const float* __restrict__ x, const __bf16* __restrict__ wp,
                    const float* __restrict__ bias, float* __restrict__ y)
{
    const int lane  = threadIdx.x & 31;
    const int wave  = threadIdx.x >> 5;
    const int group = lane >> 4;
    const int mrow  = lane & 15;

    const int tile = blockIdx.x * 8 + wave;       // 65536 tiles exactly
    const int tn   = tile & 1;
    const int tm   = tile >> 1;
    const int owb  = (tm & 31) << 4;
    const int oh   = (tm >> 5) & 511;
    const int n    = tm >> 14;

    const int ow = owb + mrow;
    const int co = tn * 16 + mrow;

    const __bf16* bptr = wp + ((size_t)tn * 32 + lane) * 16;

    v8f acc = {};
    for (int kc = 0; kc < 160; kc += 32) {        // K = 3*49 = 147, pad to 160
        const v16bf b = *(const v16bf*)bptr;
        bptr += 1024;                             // 2*32*16 elements per chunk
        v16bf a;
#pragma unroll
        for (int e = 0; e < 16; ++e) {
            int k = kc + klocal_of(e, group);
            const bool kin = (k < 147);
            if (!kin) k = 146;
            const int tap = k / 3;
            const int ci  = k - tap * 3;
            const int kh  = tap / 7;
            const int kw  = tap - kh * 7;
            const int ih  = reflect_idx(oh + kh - 3, 512);
            const int iw  = reflect_idx(ow + kw - 3, 512);
            const float av = x[(((size_t)n * 3 + ci) * 512 + ih) * 512 + iw];
            a[e] = (__bf16)(kin ? av : 0.0f);
        }
        acc = __builtin_amdgcn_wmma_f32_16x16x32_bf16(
                  false, a, false, b, (short)0, acc, false, false);
    }

    const float bval = bias[co];
    float4 lo, hi;
    lo.x = acc[0] + bval; lo.y = acc[1] + bval;
    lo.z = acc[2] + bval; lo.w = acc[3] + bval;
    hi.x = acc[4] + bval; hi.y = acc[5] + bval;
    hi.z = acc[6] + bval; hi.w = acc[7] + bval;
    float* yb = y + (((size_t)n * 32 + co) * 512 + oh) * 512 + owb + group * 8;
    *(float4*)(yb)     = lo;
    *(float4*)(yb + 4) = hi;
}

// ---------------------------------------------------------------------------
// BatchNorm stats over f32 NCHW conv output
// ---------------------------------------------------------------------------
__global__ void zero_kernel(float* p, int n) {
    int i = blockIdx.x * blockDim.x + threadIdx.x;
    if (i < n) p[i] = 0.0f;
}

__global__ __launch_bounds__(256)
void bn_stats_kernel(const float* __restrict__ x, float* __restrict__ sum,
                     float* __restrict__ sumsq, int N, int C, int HW,
                     int blocksPerC)
{
    const int c   = blockIdx.x / blocksPerC;
    const int blk = blockIdx.x % blocksPerC;
    float s = 0.0f, s2 = 0.0f;
    for (int nn = 0; nn < N; ++nn) {
        const float* p = x + ((size_t)nn * C + c) * HW;
        for (int r = blk * blockDim.x + threadIdx.x; r < HW;
             r += blocksPerC * blockDim.x) {
            const float v = p[r];
            s += v; s2 += v * v;
        }
    }
    __shared__ float ls[256], ls2[256];
    ls[threadIdx.x] = s; ls2[threadIdx.x] = s2;
    __syncthreads();
    for (int st = blockDim.x >> 1; st > 0; st >>= 1) {
        if ((int)threadIdx.x < st) {
            ls[threadIdx.x]  += ls[threadIdx.x + st];
            ls2[threadIdx.x] += ls2[threadIdx.x + st];
        }
        __syncthreads();
    }
    if (threadIdx.x == 0) {
        atomicAdd(&sum[c],   ls[0]);
        atomicAdd(&sumsq[c], ls2[0]);
    }
}

// ---------------------------------------------------------------------------
// BN normalize + ReLU + NCHW(f32) -> NHWC(bf16) transpose through LDS.
// Tile: 32 channels x 64 pixels. Per-channel scale/shift precomputed in LDS.
// ---------------------------------------------------------------------------
__global__ __launch_bounds__(256)
void bn_apply_relu_t(const float* __restrict__ y, __bf16* __restrict__ out,
                     const float* __restrict__ sum, const float* __restrict__ sumsq,
                     const float* __restrict__ g, const float* __restrict__ be,
                     int C, int HW, float invCnt, int tilesP)
{
    __shared__ __bf16 ls[64 * 33];
    __shared__ float  sc[32], sh[32];

    int bid = blockIdx.x;
    const int pb = (bid % tilesP) * 64;  bid /= tilesP;
    const int cc = C >> 5;
    const int cb = (bid % cc) * 32;
    const int n  = bid / cc;

    if (threadIdx.x < 32) {
        const int c = cb + threadIdx.x;
        const float m   = sum[c] * invCnt;
        const float var = sumsq[c] * invCnt - m * m;
        const float inv = rsqrtf(var + 1e-5f);
        const float s   = inv * g[c];
        sc[threadIdx.x] = s;
        sh[threadIdx.x] = be[c] - m * s;
    }
    __syncthreads();

#pragma unroll
    for (int j = 0; j < 8; ++j) {                 // 2048 elements / 256 threads
        const int t = threadIdx.x + 256 * j;
        const int c = t >> 6;                     // 0..31
        const int p = t & 63;                     // 0..63
        const float v0 = y[((size_t)n * C + cb + c) * HW + pb + p];
        const float v  = fmaf(v0, sc[c], sh[c]);
        ls[p * 33 + c] = (__bf16)(v > 0.0f ? v : 0.0f);
    }
    __syncthreads();
#pragma unroll
    for (int j = 0; j < 8; ++j) {
        const int t = threadIdx.x + 256 * j;
        const int p = t >> 5;                     // 0..63
        const int c = t & 31;                     // 0..31
        out[((size_t)n * HW + pb + p) * C + cb + c] = ls[p * 33 + c];
    }
}

// ---------------------------------------------------------------------------
// Instance mean (HW = 2^18): LDS-privatized segment sums -> global -> scatter
// ---------------------------------------------------------------------------
__global__ __launch_bounds__(256)
void inst_acc_kernel(const float* __restrict__ t, const int* __restrict__ inst,
                     float* __restrict__ sums, float* __restrict__ cnt, int B)
{
    __shared__ float ls[NUM_INST * 4];   // [32*3 channel sums | 32 counts]
    for (int i = threadIdx.x; i < NUM_INST * 4; i += blockDim.x) ls[i] = 0.0f;
    __syncthreads();
    const long total = (long)B * HW_OUT;
    for (long i = (long)blockIdx.x * blockDim.x + threadIdx.x; i < total;
         i += (long)gridDim.x * blockDim.x) {
        const int seg = inst[i];
        const long nn = i >> 18, r = i & (HW_OUT - 1);
        atomicAdd(&ls[seg * 3 + 0], t[((long)nn * 3 + 0) * HW_OUT + r]);
        atomicAdd(&ls[seg * 3 + 1], t[((long)nn * 3 + 1) * HW_OUT + r]);
        atomicAdd(&ls[seg * 3 + 2], t[((long)nn * 3 + 2) * HW_OUT + r]);
        atomicAdd(&ls[NUM_INST * 3 + seg], 1.0f);
    }
    __syncthreads();
    for (int i = threadIdx.x; i < NUM_INST * 4; i += blockDim.x) {
        if (i < NUM_INST * 3) atomicAdd(&sums[i], ls[i]);
        else                  atomicAdd(&cnt[i - NUM_INST * 3], ls[i]);
    }
}

__global__ __launch_bounds__(256)
void inst_scatter_kernel(const int* __restrict__ inst,
                         const float* __restrict__ sums,
                         const float* __restrict__ cnt,
                         float* __restrict__ out, int B)
{
    const long i = (long)blockIdx.x * blockDim.x + threadIdx.x;
    if (i >= (long)B * HW_OUT) return;
    const int seg = inst[i];
    const long nn = i >> 18, r = i & (HW_OUT - 1);
    const float c = fmaxf(cnt[seg], 1.0f);
    out[((long)nn * 3 + 0) * HW_OUT + r] = sums[seg * 3 + 0] / c;
    out[((long)nn * 3 + 1) * HW_OUT + r] = sums[seg * 3 + 1] / c;
    out[((long)nn * 3 + 2) * HW_OUT + r] = sums[seg * 3 + 2] / c;
}

// ---------------------------------------------------------------------------
// Host orchestration
// ---------------------------------------------------------------------------
struct Layer { const float *w, *b, *g, *be; };

extern "C" void kernel_launch(void* const* d_in, const int* in_sizes, int n_in,
                              void* d_out, int out_size, void* d_ws, size_t ws_size,
                              hipStream_t stream)
{
    (void)in_sizes; (void)n_in; (void)out_size; (void)ws_size;

    const float* x    = (const float*)d_in[0];
    const int*   inst = (const int*)d_in[1];

    int pi = 2;
    auto nextp = [&]() { return (const float*)d_in[pi++]; };
    Layer head; head.w = nextp(); head.b = nextp(); head.g = nextp(); head.be = nextp();
    Layer down[4], up[4];
    for (int i = 0; i < 4; ++i) { down[i].w = nextp(); down[i].b = nextp();
                                  down[i].g = nextp(); down[i].be = nextp(); }
    for (int i = 0; i < 4; ++i) { up[i].w = nextp(); up[i].b = nextp();
                                  up[i].g = nextp(); up[i].be = nextp(); }
    const float* tail_w = nextp();
    const float* tail_b = nextp();

    // workspace: f32 NCHW scratch (64MiB) + two bf16 NHWC buffers (32MiB each)
    const size_t ACT = (size_t)2 * 32 * 512 * 512;   // 16,777,216 elements
    float*  yscr  = (float*)d_ws;
    __bf16* bfA   = (__bf16*)(yscr + ACT);
    __bf16* bfB   = bfA + ACT;
    float*  bsum  = (float*)(bfB + ACT);
    float*  bsq   = bsum + 512;
    float*  isums = bsq + 512;
    float*  icnt  = isums + NUM_INST * 3;
    __bf16* wpack = (__bf16*)(icnt + NUM_INST);      // up to ~1.2M elems

    auto conv = [&](const __bf16* in, const Layer& L, float* out,
                    int N, int Cin, int H, int W, int Cout, int Ho, int Wo,
                    int KH, int KW, int stride, int pad, int mode, int act) {
        const int tilesN = (Cout + 15) / 16;
        const int total = KH * KW * (Cin / 32) * tilesN * 512;
        pack_w_cin32<<<(total + 255) / 256, 256, 0, stream>>>(
            L.w, wpack, Cin, Cout, KH, KW, mode == MODE_TCONV ? 1 : 0,
            tilesN, total);
        const int tiles  = (N * Ho * (Wo >> 4)) * tilesN;
        const int blocks = (tiles + 7) / 8;
        if (mode == MODE_ZERO)
            wmma_conv_cin32<MODE_ZERO><<<blocks, 256, 0, stream>>>(
                in, wpack, L.b, out, N, Cin, H, W, Cout, Ho, Wo, KH, KW, stride, pad, act);
        else if (mode == MODE_TCONV)
            wmma_conv_cin32<MODE_TCONV><<<blocks, 256, 0, stream>>>(
                in, wpack, L.b, out, N, Cin, H, W, Cout, Ho, Wo, KH, KW, stride, pad, act);
        else
            wmma_conv_cin32<MODE_REFLECT><<<blocks, 256, 0, stream>>>(
                in, wpack, L.b, out, N, Cin, H, W, Cout, Ho, Wo, KH, KW, stride, pad, act);
    };

    auto bn = [&](const float* ysrc, __bf16* dst, const Layer& L,
                  int N, int C, int HW) {
        zero_kernel<<<(2 * C + 255) / 256, 256, 0, stream>>>(bsum, 2 * C);
        const int blocksPerC = 8;
        bn_stats_kernel<<<C * blocksPerC, 256, 0, stream>>>(
            (const float*)ysrc, bsum, bsq, N, C, HW, blocksPerC);
        const int tilesP = HW / 64;
        const int blocks = N * (C / 32) * tilesP;
        bn_apply_relu_t<<<blocks, 256, 0, stream>>>(
            ysrc, dst, bsum, bsq, L.g, L.be, C, HW,
            1.0f / ((float)N * (float)HW), tilesP);
    };

    // ---- head: reflect-pad 3, 7x7, 3->32, BN+ReLU -> bf16 NHWC ----
    pack_w_head<<<(5 * 2 * 32 * 16 + 255) / 256, 256, 0, stream>>>(head.w, wpack);
    wmma_conv_head<<<65536 / 8, 256, 0, stream>>>(x, wpack, head.b, yscr);
    bn(yscr, bfA, head, 2, 32, 512 * 512);

    __bf16* cur = bfA;
    __bf16* oth = bfB;
    int Cin = 32, H = 512, W = 512;

    // ---- down path: 3x3 s2 p1, BN+ReLU ----
    for (int i = 0; i < 4; ++i) {
        const int Cout = Cin * 2, Ho = H / 2, Wo = W / 2;
        conv(cur, down[i], yscr, 2, Cin, H, W, Cout, Ho, Wo, 3, 3, 2, 1, MODE_ZERO, 0);
        bn(yscr, oth, down[i], 2, Cout, Ho * Wo);
        __bf16* t = cur; cur = oth; oth = t;
        Cin = Cout; H = Ho; W = Wo;
    }

    // ---- up path: convT(k=3,s=2,p=1,op=1) as lhs-dilated conv, BN+ReLU ----
    for (int i = 0; i < 4; ++i) {
        const int Cout = Cin / 2, Ho = H * 2, Wo = W * 2;
        conv(cur, up[i], yscr, 2, Cin, H, W, Cout, Ho, Wo, 3, 3, 1, 0, MODE_TCONV, 0);
        bn(yscr, oth, up[i], 2, Cout, Ho * Wo);
        __bf16* t = cur; cur = oth; oth = t;
        Cin = Cout; H = Ho; W = Wo;
    }

    // ---- tail: reflect-pad 3, 7x7, 32->3, tanh (no BN), f32 NCHW out ----
    {
        Layer tail; tail.w = tail_w; tail.b = tail_b; tail.g = nullptr; tail.be = nullptr;
        conv(cur, tail, yscr, 2, 32, 512, 512, 3, 512, 512, 7, 7, 1, 3,
             MODE_REFLECT, /*act=tanh*/1);
    }

    // ---- per-instance, per-channel mean over the whole batch ----
    zero_kernel<<<1, 256, 0, stream>>>(isums, NUM_INST * 4);
    inst_acc_kernel<<<512, 256, 0, stream>>>(yscr, inst, isums, icnt, 2);
    const long pix = (long)2 * HW_OUT;
    inst_scatter_kernel<<<(int)((pix + 255) / 256), 256, 0, stream>>>(
        inst, isums, icnt, (float*)d_out, 2);
}
